// MeSH_GCN_Multi_27264452395408
// MI455X (gfx1250) — compile-verified, matrix operand order checked by reading, not run on hardware
//
#include <hip/hip_runtime.h>
#include <hip/hip_fp16.h>

typedef __attribute__((ext_vector_type(16))) _Float16 v16h;
typedef __attribute__((ext_vector_type(8)))  float    v8f;

#define NLAB   28415
#define NPAD   28416      // 1776 tiles of 16
#define NT16   1776
#define EMB    200
#define KP     224        // 200 padded to 7*32
#define NKCH   7          // KP/32
#define NKF    200
#define TPA    416        // 398 padded to 13*32
#define TPT    32         // 30 padded to 32
#define HID    512
#define CTXN   1000
#define D2     400        // 2*EMB
#define K2     416        // 400 padded to 13*32
#define LA     400
#define LT     32

// ---------------- WMMA fragment helpers (ISA 7.12.2 layouts) ----------------

__device__ __forceinline__ v8f wmma16(v16h a, v16h b, v8f c) {
  return __builtin_amdgcn_wmma_f32_16x16x32_f16(false, a, false, b, (short)0, c,
                                                false, false);
}

// A fragment (16x32 f16, MxK): row-major source, lane<16 -> K {0..7,16..23},
// lane>=16 -> K {8..15,24..31}. 'row' already includes (lane&15).
__device__ __forceinline__ v16h fragA(const _Float16* A, int lda, int row,
                                      int k0, int lane) {
  const _Float16* p = A + (size_t)row * lda + k0 + ((lane & 16) ? 8 : 0);
  v16h f;
#pragma unroll
  for (int i = 0; i < 8; ++i) f[i] = p[i];
#pragma unroll
  for (int i = 0; i < 8; ++i) f[i + 8] = p[16 + i];
  return f;
}

// B fragment (32x16 f16, KxN) from row-major B[k][n]; 'col' includes (lane&15).
__device__ __forceinline__ v16h fragB(const _Float16* B, int ldb, int k0,
                                      int col, int lane) {
  const _Float16* p = B + (size_t)(k0 + ((lane & 16) ? 8 : 0)) * ldb + col;
  v16h f;
#pragma unroll
  for (int i = 0; i < 8; ++i) f[i] = p[(size_t)i * ldb];
#pragma unroll
  for (int i = 0; i < 8; ++i) f[i + 8] = p[(size_t)(16 + i) * ldb];
  return f;
}

// ---------------- small utility kernels ----------------

__global__ void zero_kernel(float4* p, long n) {
  long i = blockIdx.x * (long)blockDim.x + threadIdx.x;
  if (i < n) p[i] = make_float4(0.f, 0.f, 0.f, 0.f);
}

__global__ void cvt_f16_kernel(const float* src, _Float16* dst, long rows,
                               int scols, int dstride) {
  long i = blockIdx.x * (long)blockDim.x + threadIdx.x;
  long tot = rows * scols;
  if (i >= tot) return;
  long r = i / scols;
  int c = (int)(i - r * scols);
  dst[(size_t)r * dstride + c] = (_Float16)src[i];
}

__global__ void copy_g_to_lf_kernel(const float* g, float* LF) {
  long i = blockIdx.x * (long)blockDim.x + threadIdx.x;
  long tot = (long)NLAB * EMB;
  if (i >= tot) return;
  long r = i / EMB;
  int c = (int)(i - r * EMB);
  LF[(size_t)r * D2 + EMB + c] = g[i];
}

// ---------------- conv front-end: relu(conv1d(emb[idx])) in both f16 layouts --

__global__ void conv_kernel(const int* idx, int L, int TP, const float* emb,
                            const float* cw, const float* cb,
                            _Float16* conv_h, _Float16* convT_h) {
  int b = blockIdx.x / TP;
  int t = blockIdx.x % TP;
  int Lc = L - 2;
  __shared__ float win[3 * EMB];
  if (t < Lc) {
    for (int i = threadIdx.x; i < 3 * EMB; i += blockDim.x) {
      int w = i / EMB, e = i - w * EMB;
      win[i] = emb[(size_t)idx[b * L + t + w] * EMB + e];
    }
  }
  __syncthreads();
  int j = threadIdx.x;
  if (j < NKF) {
    float acc = 0.f;
    if (t < Lc) {
      acc = cb[j];
      for (int w = 0; w < 3; ++w)
        for (int e = 0; e < EMB; ++e)
          acc += win[w * EMB + e] * cw[(size_t)(w * EMB + e) * NKF + j];
      acc = acc > 0.f ? acc : 0.f;
    }
    conv_h[((size_t)b * TP + t) * KP + j] = (_Float16)acc;
    convT_h[((size_t)b * KP + j) * TP + t] = (_Float16)acc;
  }
}

// ---------------- GCN scatter (segment_sum) ----------------

__global__ void scatter_f32_kernel(const int* src, const int* dst, long EF,
                                   int F, const float* feat, float* out) {
  long i = blockIdx.x * (long)blockDim.x + threadIdx.x;
  if (i >= EF) return;
  long e = i / F;
  int f = (int)(i - e * F);
  atomicAdd(&out[(size_t)dst[e] * F + f], feat[(size_t)src[e] * F + f]);
}

__global__ void scatter_f16_kernel(const int* src, const int* dst, long EF,
                                   int F, const _Float16* feat, float* out) {
  long i = blockIdx.x * (long)blockDim.x + threadIdx.x;
  if (i >= EF) return;
  long e = i / F;
  int f = (int)(i - e * F);
  atomicAdd(&out[(size_t)dst[e] * F + f],
            (float)feat[(size_t)src[e] * F + f]);
}

// ---------------- generic 16x16 WMMA GEMM: C = act(A @ W^T + b) --------------
// A: [Mpad x lda] f16 row-major. W: [n][k] f16 row-major (PyTorch Linear),
// so B[k][n] = W[n][k] -> B fragment == A-style fragment of W with row=n.

__global__ __launch_bounds__(32) void gemm_nt_kernel(
    const _Float16* A, int lda, const _Float16* W, int ldb, int kchunks,
    int ntiles, int nvalid, const float* bias, int relu, float* C, int ldc,
    _Float16* Ch, int ldch) {
  int mt = blockIdx.x / ntiles;
  int nt = blockIdx.x % ntiles;
  int lane = threadIdx.x;
  int arow = mt * 16 + (lane & 15);
  int wrow = nt * 16 + (lane & 15);
  v8f acc = {};
  for (int kc = 0; kc < kchunks; ++kc) {
    v16h a = fragA(A, lda, arow, kc * 32, lane);
    v16h b = fragA(W, ldb, wrow, kc * 32, lane);
    acc = wmma16(a, b, acc);
  }
  int n = nt * 16 + (lane & 15);
  int rbase = mt * 16 + ((lane & 16) ? 8 : 0);
  float bv = (bias && n < nvalid) ? bias[n] : 0.f;
#pragma unroll
  for (int r = 0; r < 8; ++r) {
    int m = rbase + r;
    float v = acc[r] + bv;
    if (relu) v = v > 0.f ? v : 0.f;
    if (n < nvalid) {
      if (C) C[(size_t)m * ldc + n] = v;
      if (Ch) Ch[(size_t)m * ldch + n] = (_Float16)v;
    }
  }
}

// ---------------- fused attention + x_feature + logit kernel -----------------

__device__ void atten_content(const _Float16* convp, const _Float16* convTp,
                              int TP, int Lvalid, int kofs, int lane,
                              const _Float16* Gt, float* Sbuf, _Float16* Abuf,
                              _Float16* Cbuf) {
  // S[t][n] = sum_k conv[t][k] * g[n][k]
  for (int tt = 0; tt < TP / 16; ++tt) {
    v8f acc = {};
    int row = tt * 16 + (lane & 15);
    for (int kc = 0; kc < NKCH; ++kc) {
      v16h a = fragA(convp, KP, row, kc * 32, lane);
      v16h b = fragA(Gt, KP, (lane & 15), kc * 32, lane);  // B[k][n]=Gt[n][k]
      acc = wmma16(a, b, acc);
    }
    int rb = tt * 16 + ((lane & 16) ? 8 : 0);
    int nn = lane & 15;
#pragma unroll
    for (int r = 0; r < 8; ++r) Sbuf[(rb + r) * 16 + nn] = acc[r];
  }
  __syncthreads();
  // softmax over time per label column
  if (lane < 16) {
    float mx = -1e30f;
    for (int t = 0; t < Lvalid; ++t) mx = fmaxf(mx, Sbuf[t * 16 + lane]);
    float sm = 0.f;
    for (int t = 0; t < Lvalid; ++t) {
      float e = __expf(Sbuf[t * 16 + lane] - mx);
      Sbuf[t * 16 + lane] = e;
      sm += e;
    }
    float inv = 1.f / sm;
    for (int t = 0; t < TP; ++t)
      Abuf[t * 16 + lane] =
          (_Float16)((t < Lvalid) ? Sbuf[t * 16 + lane] * inv : 0.f);
  }
  __syncthreads();
  // content[k][n] = sum_t convT[k][t] * att[t][n]
  for (int kt = 0; kt < 13; ++kt) {  // 13*16 = 208 >= 200
    v8f acc = {};
    int row = kt * 16 + (lane & 15);
    for (int tc = 0; tc < TP / 32; ++tc) {
      v16h a = fragA(convTp, TP, row, tc * 32, lane);
      v16h b = fragB(Abuf, 16, tc * 32, (lane & 15), lane);
      acc = wmma16(a, b, acc);
    }
    int rb = kt * 16 + ((lane & 16) ? 8 : 0);
    int nn = lane & 15;
#pragma unroll
    for (int r = 0; r < 8; ++r) {
      int k = rb + r;
      if (k < NKF) Cbuf[(kofs + k) * 16 + nn] = (_Float16)acc[r];
    }
  }
  __syncthreads();
}

__global__ __launch_bounds__(32) void xlogit_kernel(
    const _Float16* conv_a, const _Float16* convT_a, const _Float16* conv_t,
    const _Float16* convT_t, const _Float16* g_h, const _Float16* cf_h,
    const float* cf_b, const float* LF, float* x) {
  int b = blockIdx.x / NT16;
  int nt = blockIdx.x % NT16;
  int n0 = nt * 16;
  int lane = threadIdx.x;

  __shared__ _Float16 Gt[16 * KP];
  __shared__ float Sbuf[TPA * 16];
  __shared__ _Float16 Abuf[TPA * 16];
  __shared__ _Float16 Cbuf[K2 * 16];
  __shared__ float red[32];

  for (int i = lane; i < 16 * KP; i += 32)
    Gt[i] = g_h[(size_t)(n0 + i / KP) * KP + (i % KP)];
  for (int i = lane; i < K2 * 16; i += 32) Cbuf[i] = (_Float16)0.f;
  __syncthreads();

  atten_content(conv_a + (size_t)b * TPA * KP, convT_a + (size_t)b * KP * TPA,
                TPA, LA - 2, 0, lane, Gt, Sbuf, Abuf, Cbuf);
  atten_content(conv_t + (size_t)b * TPT * KP, convT_t + (size_t)b * KP * TPT,
                TPT, LT - 2, NKF, lane, Gt, Sbuf, Abuf, Cbuf);

  // X[d][n] = cf_w @ content ; logit = sum_d tanh(X+b) * LF[n][d]
  float partial = 0.f;
  int nn = lane & 15;
  int gn = n0 + nn;
  for (int dt = 0; dt < 25; ++dt) {
    v8f acc = {};
    int row = dt * 16 + (lane & 15);
    for (int kc = 0; kc < 13; ++kc) {
      v16h a = fragA(cf_h, K2, row, kc * 32, lane);
      v16h bfr = fragB(Cbuf, 16, kc * 32, nn, lane);
      acc = wmma16(a, bfr, acc);
    }
    int rb = dt * 16 + ((lane & 16) ? 8 : 0);
#pragma unroll
    for (int r = 0; r < 8; ++r) {
      int d = rb + r;
      float v = tanhf(acc[r] + cf_b[d]);
      partial += v * LF[(size_t)gn * D2 + d];
    }
  }
  red[lane] = partial;
  __syncthreads();
  if (lane < 16 && gn < NLAB)
    x[(size_t)b * NLAB + gn] = red[lane] + red[lane + 16];
}

// ---------------- CorNet (bandwidth-bound, f32 VALU) ----------------

__global__ void cor_ctx_kernel(const float* x, const float* w_dc,
                               const float* b_dc, float* ctx) {
  int b = blockIdx.x / CTXN;
  int c = blockIdx.x % CTXN;
  const float* wr = w_dc + (size_t)c * NLAB;
  const float* xr = x + (size_t)b * NLAB;
  float acc = 0.f;
  for (int n = threadIdx.x; n < NLAB; n += blockDim.x) {
    float s = 1.f / (1.f + __expf(-xr[n]));
    acc += s * wr[n];
  }
  __shared__ float sm[256];
  sm[threadIdx.x] = acc;
  __syncthreads();
  for (int s2 = 128; s2 > 0; s2 >>= 1) {
    if (threadIdx.x < s2) sm[threadIdx.x] += sm[threadIdx.x + s2];
    __syncthreads();
  }
  if (threadIdx.x == 0) {
    float v = sm[0] + b_dc[c];
    ctx[b * CTXN + c] = v > 0.f ? v : (__expf(v) - 1.f);  // ELU(alpha=1)
  }
}

__global__ void cor_out_kernel(const float* ctx, const float* w_cd,
                               const float* b_cd, float* x, float* out) {
  long g = blockIdx.x * (long)blockDim.x + threadIdx.x;
  if (g >= (long)4 * NLAB) return;
  int b = (int)(g / NLAB);
  int n = (int)(g - (long)b * NLAB);
  const float* wr = w_cd + (size_t)n * CTXN;
  const float* cr = ctx + b * CTXN;
  float acc = b_cd[n] + x[g];
  for (int c = 0; c < CTXN; ++c) acc += cr[c] * wr[c];
  if (out)
    out[g] = 1.f / (1.f + __expf(-acc));
  else
    x[g] = acc;
}

// ---------------- launcher ----------------

static inline long cdiv(long a, long b) { return (a + b - 1) / b; }

extern "C" void kernel_launch(void* const* d_in, const int* in_sizes, int n_in,
                              void* d_out, int out_size, void* d_ws,
                              size_t ws_size, hipStream_t stream) {
  const int* input_seq = (const int*)d_in[0];
  const int* input_title = (const int*)d_in[1];
  const float* g_node = (const float*)d_in[2];
  const int* edge_src = (const int*)d_in[3];
  const int* edge_dst = (const int*)d_in[4];
  const float* emb = (const float*)d_in[5];
  const float* conv_w = (const float*)d_in[6];
  const float* conv_b = (const float*)d_in[7];
  const float* cf_w = (const float*)d_in[8];
  const float* cf_b = (const float*)d_in[9];
  const float* gcn1_w = (const float*)d_in[10];
  const float* gcn1_b = (const float*)d_in[11];
  const float* gcn2_w = (const float*)d_in[12];
  const float* gcn2_b = (const float*)d_in[13];
  const float* w_dc1 = (const float*)d_in[14];
  const float* b_dc1 = (const float*)d_in[15];
  const float* w_cd1 = (const float*)d_in[16];
  const float* b_cd1 = (const float*)d_in[17];
  const float* w_dc2 = (const float*)d_in[18];
  const float* b_dc2 = (const float*)d_in[19];
  const float* w_cd2 = (const float*)d_in[20];
  const float* b_cd2 = (const float*)d_in[21];
  long E = in_sizes[3];
  float* out = (float*)d_out;

  // workspace layout
  char* p = (char*)d_ws;
  auto take = [&](size_t bytes) {
    void* r = (void*)p;
    p += (bytes + 255) & ~(size_t)255;
    return r;
  };
  _Float16* g_h = (_Float16*)take((size_t)NPAD * KP * 2);
  _Float16* conv_a_h = (_Float16*)take((size_t)4 * TPA * KP * 2);
  _Float16* convT_a_h = (_Float16*)take((size_t)4 * KP * TPA * 2);
  _Float16* conv_t_h = (_Float16*)take((size_t)4 * TPT * KP * 2);
  _Float16* convT_t_h = (_Float16*)take((size_t)4 * KP * TPT * 2);
  _Float16* cf_h = (_Float16*)take((size_t)D2 * K2 * 2);
  _Float16* w1_h = (_Float16*)take((size_t)HID * KP * 2);
  _Float16* w2_h = (_Float16*)take((size_t)208 * HID * 2);
  float* h0 = (float*)take((size_t)NPAD * EMB * 4);
  _Float16* h0_h = (_Float16*)take((size_t)NPAD * KP * 2);
  _Float16* h1_h = (_Float16*)take((size_t)NPAD * HID * 2);
  float* h2 = (float*)take((size_t)NPAD * HID * 4);
  _Float16* h2_h = (_Float16*)take((size_t)NPAD * HID * 2);
  float* LF = (float*)take((size_t)NPAD * D2 * 4);
  float* xbuf = (float*)take((size_t)4 * NLAB * 4);
  float* ctx = (float*)take((size_t)4 * CTXN * 4);
  size_t used = (size_t)(p - (char*)d_ws);

  // 1) zero workspace (padding + atomic accumulators)
  long nf4 = (long)(used / 16);
  zero_kernel<<<cdiv(nf4, 256), 256, 0, stream>>>((float4*)d_ws, nf4);

  // 2) f16 conversions (padded)
  cvt_f16_kernel<<<cdiv((long)NLAB * EMB, 256), 256, 0, stream>>>(
      g_node, g_h, NLAB, EMB, KP);
  cvt_f16_kernel<<<cdiv((long)D2 * D2, 256), 256, 0, stream>>>(cf_w, cf_h, D2,
                                                               D2, K2);
  cvt_f16_kernel<<<cdiv((long)HID * EMB, 256), 256, 0, stream>>>(
      gcn1_w, w1_h, HID, EMB, KP);
  cvt_f16_kernel<<<cdiv((long)EMB * HID, 256), 256, 0, stream>>>(
      gcn2_w, w2_h, EMB, HID, HID);

  // 3) conv front-end (seq + title), emits [t][k] and [k][t] f16
  conv_kernel<<<4 * TPA, 256, 0, stream>>>(input_seq, LA, TPA, emb, conv_w,
                                           conv_b, conv_a_h, convT_a_h);
  conv_kernel<<<4 * TPT, 256, 0, stream>>>(input_title, LT, TPT, emb, conv_w,
                                           conv_b, conv_t_h, convT_t_h);

  // 4) GCN: scatter1 -> gemm1(relu) -> scatter2 -> gemm2 -> LF
  scatter_f32_kernel<<<cdiv(E * EMB, 256), 256, 0, stream>>>(
      edge_src, edge_dst, E * EMB, EMB, g_node, h0);
  cvt_f16_kernel<<<cdiv((long)NLAB * EMB, 256), 256, 0, stream>>>(h0, h0_h,
                                                                  NLAB, EMB, KP);
  gemm_nt_kernel<<<NT16 * (HID / 16), 32, 0, stream>>>(
      h0_h, KP, w1_h, KP, NKCH, HID / 16, HID, gcn1_b, 1, nullptr, 0, h1_h,
      HID);
  scatter_f16_kernel<<<cdiv(E * HID, 256), 256, 0, stream>>>(
      edge_src, edge_dst, E * HID, HID, h1_h, h2);
  cvt_f16_kernel<<<cdiv((long)NLAB * HID, 256), 256, 0, stream>>>(
      h2, h2_h, NLAB, HID, HID);
  gemm_nt_kernel<<<NT16 * 13, 32, 0, stream>>>(h2_h, HID, w2_h, HID, HID / 32,
                                               13, EMB, gcn2_b, 0, LF, D2,
                                               nullptr, 0);
  copy_g_to_lf_kernel<<<cdiv((long)NLAB * EMB, 256), 256, 0, stream>>>(g_node,
                                                                       LF);

  // 5) fused attention + x_feature + logits
  xlogit_kernel<<<4 * NT16, 32, 0, stream>>>(conv_a_h, convT_a_h, conv_t_h,
                                             convT_t_h, g_h, cf_h, cf_b, LF,
                                             xbuf);

  // 6) CorNet x2, final sigmoid into d_out
  cor_ctx_kernel<<<4 * CTXN, 256, 0, stream>>>(xbuf, w_dc1, b_dc1, ctx);
  cor_out_kernel<<<cdiv((long)4 * NLAB, 256), 256, 0, stream>>>(
      ctx, w_cd1, b_cd1, xbuf, nullptr);
  cor_ctx_kernel<<<4 * CTXN, 256, 0, stream>>>(xbuf, w_dc2, b_dc2, ctx);
  cor_out_kernel<<<cdiv((long)4 * NLAB, 256), 256, 0, stream>>>(
      ctx, w_cd2, b_cd2, xbuf, out);
  (void)n_in;
  (void)out_size;
  (void)ws_size;
}
